// MultiHeadAttention_85710367359164
// MI455X (gfx1250) — matrix operand order repaired and checked by reading, not compiled, not run on hardware
//
#include <hip/hip_runtime.h>
#include <hip/hip_bf16.h>
#include <math.h>

#define D_MODEL 1024
#define N_HEADS 16
#define D_K     64
#define BATCH   2
#define SEQ     2048
#define MROWS   (BATCH * SEQ)

typedef __attribute__((ext_vector_type(16))) __bf16 v16bf;
typedef __attribute__((ext_vector_type(8)))  float  v8f;

union Frag {
    unsigned int u[8];
    v16bf v;
};
union BF2 {
    __bf16 h[2];
    unsigned int u;
};

// fp32 -> packed bf16x2; native v_cvt path via fptrunc on gfx1250
static __device__ __forceinline__ unsigned int pack_bf16(float lo, float hi) {
    BF2 b;
    b.h[0] = (__bf16)lo;
    b.h[1] = (__bf16)hi;
    return b.u;
}

static __device__ __forceinline__ v8f wmma_bf16(const Frag& a, const Frag& b, v8f c) {
    return __builtin_amdgcn_wmma_f32_16x16x32_bf16(
        /*neg_a=*/false, a.v, /*neg_b=*/false, b.v,
        /*c_mod=*/(short)0, c, /*reuse_a=*/false, /*reuse_b=*/false);
}

// --- CDNA5 async global->LDS copy (ASYNCcnt tracked), GV addressing form ---
static __device__ __forceinline__ void async_copy_b128(unsigned lds_byte_off,
                                                       const void* gptr) {
    unsigned long long ga = (unsigned long long)(size_t)gptr;
    asm volatile("global_load_async_to_lds_b128 %0, %1, off"
                 :: "v"(lds_byte_off), "v"(ga)
                 : "memory");
}
static __device__ __forceinline__ void wait_async0() {
#if __has_builtin(__builtin_amdgcn_s_wait_asynccnt)
    __builtin_amdgcn_s_wait_asynccnt(0);
#else
    asm volatile("s_wait_asynccnt 0" ::: "memory");
#endif
}
static __device__ __forceinline__ unsigned lds_off(const void* p) {
    return (unsigned)(size_t)p;  // low 32 bits of flat addr = wave LDS offset
}

// ---------------------------------------------------------------------------
// Bulk fp32 -> bf16 conversion (one pass; removes conversions from hot loops)
// ---------------------------------------------------------------------------
__global__ __launch_bounds__(256)
void to_bf16_kernel(const float* __restrict__ src, unsigned int* __restrict__ dst,
                    int npairs) {
    const int i = (blockIdx.x * 256 + threadIdx.x) * 4;
#pragma unroll
    for (int j = 0; j < 4; ++j)
        dst[i + j] = pack_bf16(src[2 * (i + j)], src[2 * (i + j) + 1]);
}

// ---------------------------------------------------------------------------
// GEMM: C[m][n] = sum_k A[m][k] * W[n][k] + bias[n]  with bf16 A/W inputs.
// Block 128 (4 waves), tile 64x64, wave tile 32x32, K chunk 32.
// Staging via GLOBAL_LOAD_ASYNC_TO_LDS_B128.
// ---------------------------------------------------------------------------
template <bool OUT_BF16>
__global__ __launch_bounds__(128)
void gemm_bias_kernel(const __bf16* __restrict__ A, const __bf16* __restrict__ W,
                      const float* __restrict__ bias, void* __restrict__ Cout,
                      int Ndim, int Kdim) {
    __shared__ unsigned int shA[64][16];  // [row][k-pair] = 32 bf16 per row
    __shared__ unsigned int shB[64][16];  // [col(n)][k-pair]

    const int tid  = threadIdx.x;
    const int lane = tid & 31;
    const int wave = tid >> 5;
    const int lo16 = lane & 15;
    const int hi   = lane >> 4;

    const int m0 = blockIdx.x * 64;
    const int n0 = blockIdx.y * 64;
    const int wm = (wave >> 1) * 32;
    const int wn = (wave & 1) * 32;

    const v8f zero = {0.f, 0.f, 0.f, 0.f, 0.f, 0.f, 0.f, 0.f};
    v8f acc[2][2] = {{zero, zero}, {zero, zero}};

    // staging: 2 threads per row, each owns two 16B chunks (8 bf16 each)
    const int srow = tid >> 1;
    const int sq   = (tid & 1) * 2;  // chunk base 0 or 2
    const unsigned ldsA = lds_off(&shA[srow][0]) + sq * 16;
    const unsigned ldsB = lds_off(&shB[srow][0]) + sq * 16;
    const __bf16* gA = A + (size_t)(m0 + srow) * Kdim + sq * 8;
    const __bf16* gB = W + (size_t)(n0 + srow) * Kdim + sq * 8;

    for (int k0 = 0; k0 < Kdim; k0 += 32) {
        async_copy_b128(ldsA,      gA + k0);
        async_copy_b128(ldsA + 16, gA + k0 + 8);
        async_copy_b128(ldsB,      gB + k0);
        async_copy_b128(ldsB + 16, gB + k0 + 8);
        if (k0 + 32 < Kdim) {
            __builtin_prefetch(gA + k0 + 32, 0, 1);
            __builtin_prefetch(gB + k0 + 32, 0, 1);
        }
        wait_async0();
        __syncthreads();

        Frag afr[2], bfr[2];
#pragma unroll
        for (int mt = 0; mt < 2; ++mt) {
            const unsigned int* pa = &shA[wm + mt * 16 + lo16][0];
#pragma unroll
            for (int v = 0; v < 8; ++v)
                afr[mt].u[v] = pa[hi * 4 + (v & 3) + (v >> 2) * 8];
        }
#pragma unroll
        for (int nt = 0; nt < 2; ++nt) {
            const unsigned int* pb = &shB[wn + nt * 16 + lo16][0];
#pragma unroll
            for (int v = 0; v < 8; ++v)
                bfr[nt].u[v] = pb[hi * 4 + (v & 3) + (v >> 2) * 8];
        }
#pragma unroll
        for (int mt = 0; mt < 2; ++mt)
#pragma unroll
            for (int nt = 0; nt < 2; ++nt)
                acc[mt][nt] = wmma_bf16(afr[mt], bfr[nt], acc[mt][nt]);
        __syncthreads();
    }

#pragma unroll
    for (int mt = 0; mt < 2; ++mt)
#pragma unroll
        for (int nt = 0; nt < 2; ++nt) {
            const int n = n0 + wn + nt * 16 + lo16;
            const float bv_ = bias[n];
#pragma unroll
            for (int i = 0; i < 8; ++i) {
                const int m = m0 + wm + mt * 16 + hi * 8 + i;
                const float r = acc[mt][nt][i] + bv_;
                if (OUT_BF16)
                    ((__bf16*)Cout)[(size_t)m * Ndim + n] = (__bf16)r;
                else
                    ((float*)Cout)[(size_t)m * Ndim + n] = r;
            }
        }
}

// ---------------------------------------------------------------------------
// Causal flash attention over bf16 Q/K/V. Grid (SEQ/64, N_HEADS, BATCH), 128 thr.
// Per wave: 16 query rows, streaming 32-key tiles, online softmax in f32.
// Output written as bf16 [B*S][D_MODEL], ready for the Wo GEMM.
// ---------------------------------------------------------------------------
__global__ __launch_bounds__(128)
void attention_kernel(const __bf16* __restrict__ Q, const __bf16* __restrict__ Kp,
                      const __bf16* __restrict__ Vp, __bf16* __restrict__ O) {
    __shared__ unsigned int shQ[64][32];   // [q local][dim-pair]
    __shared__ unsigned int shK[32][32];   // [key local][dim-pair]
    __shared__ unsigned int shVT[64][16];  // [dim][key-pair] (transposed V)
    __shared__ float        shS[4][16][32];
    __shared__ unsigned int shP[4][16][16];
    __shared__ float        shStat[4][16];

    const int tid  = threadIdx.x;
    const int lane = tid & 31;
    const int wave = tid >> 5;
    const int lo16 = lane & 15;
    const int hi   = lane >> 4;

    const int q0 = blockIdx.x * 64;
    const int ho = blockIdx.y * D_K;
    const long long rowbase = (long long)blockIdx.z * SEQ;

    // --- stage Q (64x64 bf16) via async copies ---
    {
        const int r    = tid >> 1;
        const int half = tid & 1;  // 64B halves of the 128B row
        const unsigned ldsQ = lds_off(&shQ[r][0]) + half * 64;
        const __bf16* gq = Q + (size_t)(rowbase + q0 + r) * D_MODEL + ho + half * 32;
#pragma unroll
        for (int qd = 0; qd < 4; ++qd)
            async_copy_b128(ldsQ + qd * 16, gq + qd * 8);
    }
    wait_async0();
    __syncthreads();

    Frag aq[2];
#pragma unroll
    for (int c = 0; c < 2; ++c) {
        const unsigned int* p = &shQ[wave * 16 + lo16][0];
#pragma unroll
        for (int v = 0; v < 8; ++v)
            aq[c].u[v] = p[c * 16 + hi * 4 + (v & 3) + (v >> 2) * 8];
    }

    const v8f zero = {0.f, 0.f, 0.f, 0.f, 0.f, 0.f, 0.f, 0.f};
    v8f oacc[4] = {zero, zero, zero, zero};
    float m_run = -__builtin_inff();
    float l_run = 0.0f;

    const int ntiles = (q0 + 64 + 31) >> 5;
    for (int jt = 0; jt < ntiles; ++jt) {
        const int j0 = jt * 32;

        // --- stage K tile (async) + V^T tile (register transpose of bf16 bits) ---
        {
            const int key = tid >> 2;
            const int kc  = tid & 3;
            const unsigned ldsK = lds_off(&shK[key][0]) + kc * 32;
            const __bf16* gk = Kp + (size_t)(rowbase + j0 + key) * D_MODEL + ho + kc * 16;
            async_copy_b128(ldsK, gk);
            async_copy_b128(ldsK + 16, gk + 8);

            const unsigned short* gv =
                (const unsigned short*)(Vp + (size_t)(rowbase + j0 + key) * D_MODEL + ho + kc * 16);
            unsigned short* vt = (unsigned short*)shVT;
#pragma unroll
            for (int d = 0; d < 16; ++d)
                vt[(kc * 16 + d) * 32 + key] = gv[d];
        }
        wait_async0();
        __syncthreads();

        // --- scores S = Q K^T (16x32): 4 WMMAs ---
        v8f sacc[2];
#pragma unroll
        for (int nt = 0; nt < 2; ++nt) {
            Frag bk0, bk1;
            const unsigned int* pk = &shK[nt * 16 + lo16][0];
#pragma unroll
            for (int v = 0; v < 8; ++v) {
                bk0.u[v] = pk[hi * 4 + (v & 3) + (v >> 2) * 8];
                bk1.u[v] = pk[16 + hi * 4 + (v & 3) + (v >> 2) * 8];
            }
            v8f s = zero;
            s = wmma_bf16(aq[0], bk0, s);
            s = wmma_bf16(aq[1], bk1, s);
            sacc[nt] = s;
        }

        // scale + causal mask -> per-wave LDS
#pragma unroll
        for (int nt = 0; nt < 2; ++nt)
#pragma unroll
            for (int i = 0; i < 8; ++i) {
                const int qrow = q0 + wave * 16 + hi * 8 + i;
                const int key  = j0 + nt * 16 + lo16;
                float sv = sacc[nt][i] * 0.125f;  // 1/sqrt(64)
                if (key > qrow) sv = -__builtin_inff();
                shS[wave][hi * 8 + i][nt * 16 + lo16] = sv;
            }
        __syncthreads();

        // --- online softmax: lane r < 16 owns row r of this wave ---
        if (lane < 16) {
            const int r = lane;
            float sv[32];
            float mrow = -__builtin_inff();
#pragma unroll
            for (int k = 0; k < 32; ++k) {
                sv[k] = shS[wave][r][k];
                mrow = fmaxf(mrow, sv[k]);
            }
            const float m_new = fmaxf(m_run, mrow);
            const float corr  = __expf(m_run - m_new);
            float lsum = 0.0f;
#pragma unroll
            for (int k = 0; k < 16; ++k) {
                const float p0 = __expf(sv[2 * k]     - m_new);
                const float p1 = __expf(sv[2 * k + 1] - m_new);
                lsum += p0 + p1;
                shP[wave][r][k] = pack_bf16(p0, p1);
            }
            l_run = l_run * corr + lsum;
            m_run = m_new;
            shStat[wave][r] = corr;
        }
        __syncthreads();

        // --- rescale accumulators, O += P @ V (4 WMMAs) ---
        {
            const float* ps = &shStat[wave][hi * 8];
            v8f cv;
#pragma unroll
            for (int i = 0; i < 8; ++i) cv[i] = ps[i];

            Frag ap;
            const unsigned int* pp = &shP[wave][lo16][0];
#pragma unroll
            for (int v = 0; v < 8; ++v)
                ap.u[v] = pp[hi * 4 + (v & 3) + (v >> 2) * 8];

#pragma unroll
            for (int nt = 0; nt < 4; ++nt) {
                Frag bv_;
                const unsigned int* pv = &shVT[nt * 16 + lo16][0];
#pragma unroll
                for (int v = 0; v < 8; ++v) bv_.u[v] = pv[hi * 4 + (v & 3) + (v >> 2) * 8];
                oacc[nt] = wmma_bf16(ap, bv_, oacc[nt] * cv);
            }
        }
        __syncthreads();
    }

    // --- finalize: divide by row sums, store bf16 ---
    if (lane < 16) shStat[wave][lane] = 1.0f / l_run;
    __syncthreads();
    {
        const float* ps = &shStat[wave][hi * 8];
        v8f cv;
#pragma unroll
        for (int i = 0; i < 8; ++i) cv[i] = ps[i];
#pragma unroll
        for (int nt = 0; nt < 4; ++nt) {
            const v8f ov = oacc[nt] * cv;
#pragma unroll
            for (int i = 0; i < 8; ++i) {
                const size_t m = (size_t)(rowbase + q0 + wave * 16 + hi * 8 + i);
                O[m * D_MODEL + ho + nt * 16 + lo16] = (__bf16)ov[i];
            }
        }
    }
}

// ---------------------------------------------------------------------------
extern "C" void kernel_launch(void* const* d_in, const int* in_sizes, int n_in,
                              void* d_out, int out_size, void* d_ws, size_t ws_size,
                              hipStream_t stream) {
    (void)in_sizes; (void)n_in; (void)out_size; (void)ws_size;

    const float* x  = (const float*)d_in[0];
    // d_in[1] causal mask: applied analytically in-kernel.
    const float* Wq = (const float*)d_in[2];
    const float* bq = (const float*)d_in[3];
    const float* Wk = (const float*)d_in[4];
    const float* bk = (const float*)d_in[5];
    const float* Wv = (const float*)d_in[6];
    const float* bv = (const float*)d_in[7];
    const float* Wo = (const float*)d_in[8];
    const float* bo = (const float*)d_in[9];
    float* out = (float*)d_out;

    const size_t XE = (size_t)MROWS * D_MODEL;    // 4M elements
    const size_t WE = (size_t)D_MODEL * D_MODEL;  // 1M elements

    __bf16* xbf  = (__bf16*)d_ws;
    __bf16* Wqbf = xbf + XE;
    __bf16* Wkbf = Wqbf + WE;
    __bf16* Wvbf = Wkbf + WE;
    __bf16* Wobf = Wvbf + WE;
    __bf16* Qbf  = Wobf + WE;
    __bf16* Kbf  = Qbf + XE;
    __bf16* Vbf  = Kbf + XE;
    __bf16* Abf  = Vbf + XE;

    const int XP = (int)(XE / 2);  // bf16 pairs
    const int WP = (int)(WE / 2);
    to_bf16_kernel<<<XP / 1024, 256, 0, stream>>>(x,  (unsigned int*)xbf,  XP);
    to_bf16_kernel<<<WP / 1024, 256, 0, stream>>>(Wq, (unsigned int*)Wqbf, WP);
    to_bf16_kernel<<<WP / 1024, 256, 0, stream>>>(Wk, (unsigned int*)Wkbf, WP);
    to_bf16_kernel<<<WP / 1024, 256, 0, stream>>>(Wv, (unsigned int*)Wvbf, WP);
    to_bf16_kernel<<<WP / 1024, 256, 0, stream>>>(Wo, (unsigned int*)Wobf, WP);

    dim3 gg(MROWS / 64, D_MODEL / 64);
    gemm_bias_kernel<true><<<gg, 128, 0, stream>>>(xbf, Wqbf, bq, Qbf, D_MODEL, D_MODEL);
    gemm_bias_kernel<true><<<gg, 128, 0, stream>>>(xbf, Wkbf, bk, Kbf, D_MODEL, D_MODEL);
    gemm_bias_kernel<true><<<gg, 128, 0, stream>>>(xbf, Wvbf, bv, Vbf, D_MODEL, D_MODEL);

    dim3 ga(SEQ / 64, N_HEADS, BATCH);
    attention_kernel<<<ga, 128, 0, stream>>>(Qbf, Kbf, Vbf, Abf);

    gemm_bias_kernel<false><<<gg, 128, 0, stream>>>(Abf, Wobf, bo, out, D_MODEL, D_MODEL);
}
